// ConvAttention_36223754174593
// MI455X (gfx1250) — matrix operand order repaired and checked by reading, not compiled, not run on hardware
//
#include <hip/hip_runtime.h>
#include <hip/hip_bf16.h>
#include <math.h>

typedef __attribute__((ext_vector_type(16))) _Float16 v16h;
typedef __attribute__((ext_vector_type(8)))  float    v8f;

#define NUM_HEADS 8
#define DIM_HEAD  64
#define ATTN_DIM  512
#define NPIX      1024   // H*W = 32*32
#define BATCH     8
#define C_IN      256
#define C_OUT     256
#define QKV_COLS  1536
#define QK_SCALE  0.125f // 64^-0.5
#define GN_EPS    1e-5f

// D = A(16x32 f16) * B(32x16 f16) + C(16x16 f32)
__device__ __forceinline__ v8f wmma16(v16h a, v16h b, v8f c) {
  return __builtin_amdgcn_wmma_f32_16x16x32_f16(false, a, false, b,
                                                (short)0, c, false, false);
}

// 16-bit A 16x32 k order used here (contiguous): element e (0..15) of lane (half,l16) holds
// K = half*16 + e
__device__ __forceinline__ int a_krel(int half, int e) {
  return half * 16 + e;
}
// inverse: given K-offset krel (0..31) -> (lane-half, element index)
__device__ __forceinline__ void a_inv(int krel, int& hh, int& e) {
  hh = krel >> 4; e = krel & 15;
}
__device__ __forceinline__ int a_kbase(int half, int v) {
  return half * 16 + 2 * v;
}

// raw 32-bit LDS byte offset of a __shared__ pointer (ptrtoint of as(3) ptr)
typedef __attribute__((address_space(3))) _Float16 lds_f16;
__device__ __forceinline__ unsigned to_lds_off(_Float16* p) {
  return (unsigned)(uintptr_t)(lds_f16*)p;
}

// ---------------------------------------------------------------------------
// Fragment-swizzle kernels: pack operands so a wave loads a whole fragment as
// one contiguous v16h per lane:  buf[block*512 + lane*16 + elem]
// ---------------------------------------------------------------------------
// X (8192x256 f32) -> A-fragment f16 blocks (rowblock rb[512], kchunk kc[8])
__global__ __launch_bounds__(256) void swz_x(const float* __restrict__ X,
                                             _Float16* __restrict__ Xs) {
  const int o = blockIdx.x * 256 + threadIdx.x;   // 2,097,152
  const int e = o & 15, lane = (o >> 4) & 31, kc = (o >> 9) & 7, rb = o >> 12;
  const int half = lane >> 4, l16 = lane & 15;
  const int row = rb * 16 + l16, c = kc * 32 + a_krel(half, e);
  Xs[o] = (_Float16)X[row * C_IN + c];
}
// Wqkv (256x1536 f32) -> B-fragment f16 blocks (kc[8], colblock cb[96])
// B 32x16 packing used here (contiguous k): lane (half,l16): col=l16, elem j: k = half*16 + j
__global__ __launch_bounds__(256) void swz_wqkv(const float* __restrict__ W,
                                                _Float16* __restrict__ Ws) {
  const int o = blockIdx.x * 256 + threadIdx.x;   // 393,216
  const int j = o & 15, lane = (o >> 4) & 31, cb = (o >> 9) % 96, kc = o / 49152;
  const int half = lane >> 4, l16 = lane & 15;
  Ws[o] = (_Float16)W[(kc * 32 + half * 16 + j) * QKV_COLS + cb * 16 + l16];
}
// Wout (512x256 f32) -> B-fragment f16 blocks (kc[16], cb[16])
__global__ __launch_bounds__(256) void swz_wout(const float* __restrict__ W,
                                                _Float16* __restrict__ Ws) {
  const int o = blockIdx.x * 256 + threadIdx.x;   // 131,072
  const int j = o & 15, lane = (o >> 4) & 31, cb = (o >> 9) & 15, kc = o >> 13;
  const int half = lane >> 4, l16 = lane & 15;
  Ws[o] = (_Float16)W[(kc * 32 + half * 16 + j) * C_OUT + cb * 16 + l16];
}

// ---------------------------------------------------------------------------
// Kernel 1: QKV projection, 16x64 tile per wave (4 wmma per k-step, A reused).
// Epilogue scatters into the attention kernel's fragment layouts:
//   Qs: A-layout blocks (bh, qb[64], cc[2]), pre-scaled by dh^-0.5
//   Ks: B-layout blocks (bh, keyblk16[64], cc[2])
//   Vs: B-layout blocks (bh, keychunk32[32], ctile[4])
// ---------------------------------------------------------------------------
__global__ __launch_bounds__(32) void qkv_gemm(const _Float16* __restrict__ Xs,
                                               const _Float16* __restrict__ Ws,
                                               _Float16* __restrict__ Q,
                                               _Float16* __restrict__ Kv,
                                               _Float16* __restrict__ V) {
  const int lane = threadIdx.x & 31;
  const int half = lane >> 4, l16 = lane & 15;
  const int rb = blockIdx.x / 24, cg = blockIdx.x % 24;   // 512 x 24

  v8f acc[4] = {};
  for (int kc = 0; kc < 8; ++kc) {
    const v16h a = *(const v16h*)(Xs + ((size_t)rb * 8 + kc) * 512 + lane * 16);
    if (kc + 1 < 8)
      __builtin_prefetch(Xs + ((size_t)rb * 8 + kc + 1) * 512 + lane * 16, 0, 1);
#pragma unroll
    for (int t = 0; t < 4; ++t) {
      const int cb = cg * 4 + t;
      const v16h b = *(const v16h*)(Ws + ((size_t)kc * 96 + cb) * 512 + lane * 16);
      acc[t] = wmma16(a, b, acc[t]);
    }
  }
#pragma unroll
  for (int t = 0; t < 4; ++t) {
    const int col = (cg * 4 + t) * 16 + l16;
#pragma unroll
    for (int r = 0; r < 8; ++r) {
      const int orow = rb * 16 + half * 8 + r;      // 0..8191
      const int bb = orow >> 10, n = orow & 1023;
      const float val = acc[t][r];
      if (col < ATTN_DIM) {
        const int hd = col >> 6, c = col & 63;
        const size_t bh = (size_t)bb * 8 + hd;
        int hh, e; a_inv(c & 31, hh, e);
        Q[((bh * 64 + (n >> 4)) * 2 + (c >> 5)) * 512 + (hh * 16 + (n & 15)) * 16 + e] =
            (_Float16)(val * QK_SCALE);
      } else if (col < 2 * ATTN_DIM) {
        const int cc0 = col - ATTN_DIM, hd = cc0 >> 6, c = cc0 & 63;
        const size_t bh = (size_t)bb * 8 + hd;
        Kv[((bh * 64 + (n >> 4)) * 2 + (c >> 5)) * 512 +
           (((c & 31) >> 4) * 16 + (n & 15)) * 16 + (c & 15)] = (_Float16)val;
      } else {
        const int cc0 = col - 2 * ATTN_DIM, hd = cc0 >> 6, c = cc0 & 63;
        const size_t bh = (size_t)bb * 8 + hd;
        V[((bh * 32 + (n >> 5)) * 4 + (c >> 4)) * 512 +
          (((n & 31) >> 4) * 16 + (c & 15)) * 16 + (n & 15)] = (_Float16)val;
      }
    }
  }
}

// ---------------------------------------------------------------------------
// Kernel 2: flash attention, one wave per (b,h) x 16-query block.
// K/V chunks are prefetched one chunk ahead into an LDS double buffer with
// GLOBAL_LOAD_ASYNC_TO_LDS_B128 (ASYNCcnt-pipelined); fragments are then read
// from LDS as contiguous v16h (ds_load_b128 pairs).
// ---------------------------------------------------------------------------
__global__ __launch_bounds__(32) void attn_kernel(const _Float16* __restrict__ Qs,
                                                  const _Float16* __restrict__ Ks,
                                                  const _Float16* __restrict__ Vs,
                                                  _Float16* __restrict__ Y) {
  __shared__ _Float16 KV[2][4096];   // per buffer: K frags [0,2048) V frags [2048,4096)
  __shared__ _Float16 Pl[16 * 32];   // P tile staging for C-layout -> A-layout
  const int lane = threadIdx.x & 31;
  const int half = lane >> 4, l16 = lane & 15;
  const int bh = blockIdx.x >> 6, qb = blockIdx.x & 63;
  const int bb = bh >> 3, h = bh & 7;

  const unsigned kvbase = to_lds_off(&KV[0][0]);

  // issue async copy of one 8KB K+V chunk into LDS buffer `buf`
  auto issue = [&](int kc, int buf) {
    const _Float16* kg = Ks + ((size_t)bh * 64 + kc * 2) * 1024;
    const _Float16* vg = Vs + ((size_t)bh * 32 + kc) * 2048;
    const unsigned lk = kvbase + (unsigned)buf * 8192u + (unsigned)lane * 16u;
    const unsigned lv = lk + 4096u;
    const uint64_t gk = (uint64_t)(uintptr_t)kg + (unsigned)lane * 16u;
    const uint64_t gv = (uint64_t)(uintptr_t)vg + (unsigned)lane * 16u;
#pragma unroll
    for (int i = 0; i < 8; ++i)   // 8 x 512B = 4KB of K fragments
      asm volatile("global_load_async_to_lds_b128 %0, %1, off offset:%2"
                   :: "v"(lk), "v"(gk), "i"(i * 512) : "memory");
#pragma unroll
    for (int i = 0; i < 8; ++i)   // 8 x 512B = 4KB of V fragments
      asm volatile("global_load_async_to_lds_b128 %0, %1, off offset:%2"
                   :: "v"(lv), "v"(gv), "i"(i * 512) : "memory");
  };

  const _Float16* qp = Qs + ((size_t)bh * 64 + qb) * 1024;
  const v16h aq0 = *(const v16h*)(qp + lane * 16);          // c 0..31
  const v16h aq1 = *(const v16h*)(qp + 512 + lane * 16);    // c 32..63

  float mrun[8], lrun[8];
#pragma unroll
  for (int r = 0; r < 8; ++r) { mrun[r] = -1e30f; lrun[r] = 0.f; }
  v8f oacc[4] = {};

  issue(0, 0);                               // prologue: prefetch chunk 0
  for (int kc = 0; kc < 32; ++kc) {          // 32 chunks of 32 keys
    const int buf = kc & 1;
    if (kc + 1 < 32) {
      issue(kc + 1, buf ^ 1);                // prefetch next chunk
      // wait until only the 16 just-issued ops remain -> chunk kc is resident
      asm volatile("s_wait_asynccnt 0x10" ::: "memory");
    } else {
      asm volatile("s_wait_asynccnt 0x0" ::: "memory");
    }
    const _Float16* kb = &KV[buf][0];
    const _Float16* vb = &KV[buf][2048];
    const v16h bk0  = *(const v16h*)(kb + lane * 16);           // keys lo, c lo
    const v16h bk1  = *(const v16h*)(kb + 512 + lane * 16);     // keys lo, c hi
    const v16h bk0h = *(const v16h*)(kb + 1024 + lane * 16);    // keys hi, c lo
    const v16h bk1h = *(const v16h*)(kb + 1536 + lane * 16);    // keys hi, c hi

    v8f zero = {};
    v8f s0 = wmma16(aq0, bk0, zero);  s0 = wmma16(aq1, bk1, s0);
    v8f s1 = wmma16(aq0, bk0h, zero); s1 = wmma16(aq1, bk1h, s1);

    // online softmax; row = half*8 + r, cols live across the half's 16 lanes
    float alpha[8];
#pragma unroll
    for (int r = 0; r < 8; ++r) {
      float rm = fmaxf(s0[r], s1[r]);
#pragma unroll
      for (int off = 8; off >= 1; off >>= 1) rm = fmaxf(rm, __shfl_xor(rm, off, 32));
      const float mnew = fmaxf(mrun[r], rm);
      alpha[r] = __expf(mrun[r] - mnew);
      const float p0 = __expf(s0[r] - mnew);
      const float p1 = __expf(s1[r] - mnew);
      float rs = p0 + p1;
#pragma unroll
      for (int off = 8; off >= 1; off >>= 1) rs += __shfl_xor(rs, off, 32);
      lrun[r] = lrun[r] * alpha[r] + rs;
      mrun[r] = mnew;
      const int prow = half * 8 + r;
      Pl[prow * 32 + l16]      = (_Float16)p0;
      Pl[prow * 32 + 16 + l16] = (_Float16)p1;
    }
    // DS ops are wave-ordered; fence stops compiler reordering the cross-lane dep
    asm volatile("s_wait_dscnt 0x0" ::: "memory");

    v16h ap{};
#pragma unroll
    for (int v = 0; v < 8; ++v) {
      const int kbi = a_kbase(half, v);
      ap[2 * v]     = Pl[l16 * 32 + kbi];
      ap[2 * v + 1] = Pl[l16 * 32 + kbi + 1];
    }
#pragma unroll
    for (int t = 0; t < 4; ++t) {
      v8f c = oacc[t];
#pragma unroll
      for (int r = 0; r < 8; ++r) c[r] *= alpha[r];
      const v16h bv = *(const v16h*)(vb + t * 512 + lane * 16);
      oacc[t] = wmma16(ap, bv, c);
    }
    asm volatile("" ::: "memory");   // keep next iter's DS stores below our DS loads
  }

  // epilogue: normalize; faithful-TF reshape: out(b,h,n,c) -> (p,ch) with
  // p=(n%16)*64+c, ch=h*64+qb; written directly in proj_gemm's A-frag layout.
  const int ch = h * 64 + qb;
  const int kcp = ch >> 5;
  int hh, e; a_inv(ch & 31, hh, e);
#pragma unroll
  for (int t = 0; t < 4; ++t) {
#pragma unroll
    for (int r = 0; r < 8; ++r) {
      const int m = half * 8 + r;
      const float val = oacc[t][r] / lrun[r];
      const int rbp = m * 4 + t;               // p >> 4  (p = m*64 + t*16 + l16)
      Y[(((size_t)bb * 64 + rbp) * 16 + kcp) * 512 + (hh * 16 + l16) * 16 + e] =
          (_Float16)val;
    }
  }
}

// ---------------------------------------------------------------------------
// Kernel 3: output projection, 16x64 tile per wave, fragment loads only.
// ---------------------------------------------------------------------------
__global__ __launch_bounds__(32) void proj_gemm(const _Float16* __restrict__ Ys,
                                                const _Float16* __restrict__ Wo,
                                                const float* __restrict__ bias,
                                                float* __restrict__ Z) {
  const int lane = threadIdx.x & 31;
  const int half = lane >> 4, l16 = lane & 15;
  const int bb = blockIdx.x >> 8;                     // 8 * 64 * 4 waves
  const int rem = blockIdx.x & 255;
  const int rb = rem >> 2, cg = rem & 3;
  const _Float16* yp = Ys + ((size_t)bb * 64 + rb) * 16 * 512;

  v8f acc[4] = {};
  for (int kc = 0; kc < 16; ++kc) {
    const v16h a = *(const v16h*)(yp + kc * 512 + lane * 16);
#pragma unroll
    for (int t = 0; t < 4; ++t) {
      const int cb = cg * 4 + t;
      const v16h b = *(const v16h*)(Wo + ((size_t)kc * 16 + cb) * 512 + lane * 16);
      acc[t] = wmma16(a, b, acc[t]);
    }
  }
#pragma unroll
  for (int t = 0; t < 4; ++t) {
    const int col = (cg * 4 + t) * 16 + l16;
    const float bvv = bias[col];
#pragma unroll
    for (int r = 0; r < 8; ++r) {
      const int orow = rb * 16 + half * 8 + r;
      Z[((size_t)bb * NPIX + orow) * C_OUT + col] = acc[t][r] + bvv;
    }
  }
}

// ---------------------------------------------------------------------------
// Kernel 4: per-sample GroupNorm(groups=1) statistics over 262144 elements
// ---------------------------------------------------------------------------
__global__ __launch_bounds__(256) void gn_stats(const float* __restrict__ Z,
                                                float* __restrict__ stats) {
  __shared__ float ssum[256], ssq[256];
  const int bb = blockIdx.x;
  const float* zb = Z + (size_t)bb * (NPIX * C_OUT);
  float s = 0.f, q = 0.f;
  for (int i = threadIdx.x; i < NPIX * C_OUT; i += 256) {
    const float v = zb[i];
    s += v; q += v * v;
  }
  ssum[threadIdx.x] = s; ssq[threadIdx.x] = q;
  __syncthreads();
  for (int off = 128; off > 0; off >>= 1) {
    if (threadIdx.x < off) {
      ssum[threadIdx.x] += ssum[threadIdx.x + off];
      ssq[threadIdx.x]  += ssq[threadIdx.x + off];
    }
    __syncthreads();
  }
  if (threadIdx.x == 0) {
    const float inv = 1.0f / (float)(NPIX * C_OUT);
    const float mean = ssum[0] * inv;
    const float var  = ssq[0] * inv - mean * mean;
    stats[bb * 2]     = mean;
    stats[bb * 2 + 1] = rsqrtf(var + GN_EPS);
  }
}

// ---------------------------------------------------------------------------
// Kernel 5: apply normalization + affine
// ---------------------------------------------------------------------------
__global__ __launch_bounds__(256) void gn_apply(const float* __restrict__ Z,
                                                const float* __restrict__ stats,
                                                const float* __restrict__ gamma,
                                                const float* __restrict__ beta,
                                                float* __restrict__ out) {
  const int idx = blockIdx.x * 256 + threadIdx.x;   // 2,097,152 total
  const int bb = idx >> 18;
  const int ch = idx & 255;
  const float mean = stats[bb * 2], rstd = stats[bb * 2 + 1];
  out[idx] = (Z[idx] - mean) * rstd * gamma[ch] + beta[ch];
}

// ---------------------------------------------------------------------------
extern "C" void kernel_launch(void* const* d_in, const int* in_sizes, int n_in,
                              void* d_out, int out_size, void* d_ws, size_t ws_size,
                              hipStream_t stream) {
  (void)in_sizes; (void)n_in; (void)out_size; (void)ws_size;
  const float* x     = (const float*)d_in[0];
  const float* w_qkv = (const float*)d_in[1];
  const float* w_out = (const float*)d_in[2];
  const float* b_out = (const float*)d_in[3];
  const float* gamma = (const float*)d_in[4];
  const float* beta  = (const float*)d_in[5];
  float* out = (float*)d_out;

  // ws layout (bytes):
  //   Qs [0,8M)  Ks [8M,16M)  Vs [16M,24M)  Y-region [24M,32M)  WoutSwz [32M,32.25M)
  //   Xswz/WqkvSwz live inside the Y region (dead once attn writes Y)
  //   Z (f32) reuses [0,8M) after attention; stats at [8M,8M+64)
  char* ws = (char*)d_ws;
  _Float16* Qs = (_Float16*)(ws);
  _Float16* Ks = (_Float16*)(ws + 8388608);
  _Float16* Vs = (_Float16*)(ws + 16777216);
  _Float16* Ys = (_Float16*)(ws + 25165824);
  _Float16* Xs = (_Float16*)(ws + 25165824);             // overlaps Ys (pre-attn)
  _Float16* Wq = (_Float16*)(ws + 25165824 + 4194304);   // overlaps Ys (pre-attn)
  _Float16* Wo = (_Float16*)(ws + 33554432);
  float* Z     = (float*)(ws);
  float* stats = (float*)(ws + 8388608);

  swz_x     <<<8192, 256, 0, stream>>>(x, Xs);
  swz_wqkv  <<<1536, 256, 0, stream>>>(w_qkv, Wq);
  swz_wout  <<<512,  256, 0, stream>>>(w_out, Wo);
  qkv_gemm  <<<12288, 32, 0, stream>>>(Xs, Wq, Qs, Ks, Vs);
  attn_kernel<<<4096, 32, 0, stream>>>(Qs, Ks, Vs, Ys);
  proj_gemm <<<2048,  32, 0, stream>>>(Ys, Wo, b_out, Z);
  gn_stats  <<<8,    256, 0, stream>>>(Z, stats);
  gn_apply  <<<8192, 256, 0, stream>>>(Z, stats, gamma, beta, out);
}
